// MQNet_54271206752664
// MI455X (gfx1250) — matrix operand order repaired and checked by reading, not compiled, not run on hardware
//
#include <hip/hip_runtime.h>

// ---------------------------------------------------------------------------
// MQNet on MI455X (gfx1250): GCN (bf16 WMMA) + loop-invariant GRU input GEMM
// + lightweight sequential argmax chain, all device-side (graph capturable).
// ---------------------------------------------------------------------------

typedef __attribute__((ext_vector_type(16))) __bf16 v16bf;
typedef __attribute__((ext_vector_type(2)))  __bf16 v2bf;
typedef __attribute__((ext_vector_type(8)))  float  v8f;
typedef unsigned short u16;
typedef unsigned int   u32;

#define NN   8192
#define DD   512
#define HH   512
#define H3   1536
#define BSZ  20

// ----- CDNA5 async global->LDS (guarded: falls back to VGPR staging) -------
#if __has_builtin(__builtin_amdgcn_global_load_async_to_lds_b128)
#define HAVE_ASYNC_LDS 1
typedef int vi4 __attribute__((vector_size(16)));         // GCC vector int4
typedef __attribute__((address_space(1))) vi4 g_vi4;      // global int4
typedef __attribute__((address_space(3))) vi4 l_vi4;      // LDS int4
#else
#define HAVE_ASYNC_LDS 0
#endif

__device__ __forceinline__ void wait_async0() {
#if __has_builtin(__builtin_amdgcn_s_wait_asynccnt)
    __builtin_amdgcn_s_wait_asynccnt(0);
#elif HAVE_ASYNC_LDS
    asm volatile("s_wait_asynccnt 0x0" ::: "memory");
#endif
}

// ------------------------------- helpers -----------------------------------

__device__ __forceinline__ u16 f2bf(float f) {
    __bf16 b = (__bf16)f;                 // native v_cvt (RNE)
    return __builtin_bit_cast(u16, b);
}
__device__ __forceinline__ u32 pack2(float lo, float hi) {
    v2bf v; v.x = (__bf16)lo; v.y = (__bf16)hi;     // -> v_cvt_pk_bf16_f32
    return __builtin_bit_cast(u32, v);
}
__device__ __forceinline__ float sigm(float x) { return 1.0f / (1.0f + expf(-x)); }

__device__ __forceinline__ float wave_sum(float s) {
#pragma unroll
    for (int o = 16; o > 0; o >>= 1) s += __shfl_xor(s, o, 32);
    return s;
}

// ------------------------------ GEMM (WMMA) --------------------------------
// C[M,N] = act( A[M,K] @ B[K,N] + bias )
// A: fp32 row-major (converted in-register) OR bf16 row-major (one non-null).
// B: supplied transposed Bt[N,K], bf16 row-major (contiguous K per column).
// Outputs: optional fp32 row-major Cf, optional bf16 row-major Cb.
// Block tile 128x128, K-step 32, 8 waves; wave tile 32x64 (8 WMMA / k-step).
// Double-buffered LDS, one barrier per k-step; bf16 tiles move via
// GLOBAL_LOAD_ASYNC_TO_LDS_B128 (ASYNCcnt), fp32 adj converts through VGPRs.

#define BM 128
#define BN 128
#define BK 32
#define LDS_STRIDE 40   // bf16 elements; 80B rows -> conflict-free b128 reads

__global__ __launch_bounds__(256)
void gemm_bf16_wmma(const float* __restrict__ Af32,
                    const u16*  __restrict__ Abf,
                    const u16*  __restrict__ Bt,
                    float* __restrict__ Cf, u16* __restrict__ Cb,
                    const float* __restrict__ bias, int relu,
                    int Mtot, int Ntot, int Ktot) {
    __shared__ __align__(16) u16 As[2][BM * LDS_STRIDE];
    __shared__ __align__(16) u16 Bs[2][BN * LDS_STRIDE];

    const int tid  = threadIdx.x;
    const int wave = tid >> 5;
    const int lane = tid & 31;
    const int half = lane >> 4;       // 0: lanes 0-15, 1: lanes 16-31
    const int l16  = lane & 15;

    const int m0 = blockIdx.y * BM;
    const int n0 = blockIdx.x * BN;

    // wave tile: rows (wave>>1)*32, cols (wave&1)*64
    const int wm = (wave >> 1) * 32;
    const int wn = (wave & 1) * 64;

    v8f acc[2][4] = {};

    // A staging: thread -> row r = tid/2 (0..127), 16 elems at col (tid&1)*16
    const int ar = tid >> 1;
    const int ac = (tid & 1) * 16;
    // B staging: thread -> n = tid/2 (0..127), 16 elems at k (tid&1)*16
    const int bn_ld = tid >> 1;
    const int bk_ld = (tid & 1) * 16;

    float4 fA[4];        // raw fp32 A staging (conversion deferred past compute)
#if !HAVE_ASYNC_LDS
    uint4  uA[2];        // bf16 A staging (fallback)
    uint4  uB[2];        // bf16 B staging (fallback)
#endif

    auto loadTiles = [&](int k0, int buf) {
        // ---- B tile (always bf16): async straight into LDS ----
        const u16* sb = Bt + (size_t)(n0 + bn_ld) * Ktot + k0 + bk_ld;
        u16* db = Bs[buf] + bn_ld * LDS_STRIDE + bk_ld;
#if HAVE_ASYNC_LDS
        __builtin_amdgcn_global_load_async_to_lds_b128(
            (g_vi4*)sb, (l_vi4*)db, 0, 0);
        __builtin_amdgcn_global_load_async_to_lds_b128(
            (g_vi4*)(sb + 8), (l_vi4*)(db + 8), 0, 0);
#else
        const uint4* sbv = (const uint4*)sb;
        uB[0] = sbv[0]; uB[1] = sbv[1];
#endif
        // ---- A tile ----
        if (Af32) {
            const float4* s =
                (const float4*)(Af32 + (size_t)(m0 + ar) * Ktot + k0 + ac);
            fA[0] = s[0]; fA[1] = s[1]; fA[2] = s[2]; fA[3] = s[3];
            if (k0 + BK < Ktot)
                __builtin_prefetch(Af32 + (size_t)(m0 + ar) * Ktot + k0 + BK + ac, 0, 1);
        } else {
            const u16* sa = Abf + (size_t)(m0 + ar) * Ktot + k0 + ac;
            u16* da = As[buf] + ar * LDS_STRIDE + ac;
#if HAVE_ASYNC_LDS
            __builtin_amdgcn_global_load_async_to_lds_b128(
                (g_vi4*)sa, (l_vi4*)da, 0, 0);
            __builtin_amdgcn_global_load_async_to_lds_b128(
                (g_vi4*)(sa + 8), (l_vi4*)(da + 8), 0, 0);
#else
            const uint4* sav = (const uint4*)sa;
            uA[0] = sav[0]; uA[1] = sav[1];
#endif
        }
    };
    auto storeTiles = [&](int buf) {
        if (Af32) {   // fp32 -> packed bf16 -> LDS
            uint4 o0, o1;
            o0.x = pack2(fA[0].x, fA[0].y); o0.y = pack2(fA[0].z, fA[0].w);
            o0.z = pack2(fA[1].x, fA[1].y); o0.w = pack2(fA[1].z, fA[1].w);
            o1.x = pack2(fA[2].x, fA[2].y); o1.y = pack2(fA[2].z, fA[2].w);
            o1.z = pack2(fA[3].x, fA[3].y); o1.w = pack2(fA[3].z, fA[3].w);
            uint4* da = (uint4*)(As[buf] + ar * LDS_STRIDE + ac);
            da[0] = o0; da[1] = o1;
        }
#if !HAVE_ASYNC_LDS
        else {
            uint4* da = (uint4*)(As[buf] + ar * LDS_STRIDE + ac);
            da[0] = uA[0]; da[1] = uA[1];
        }
        uint4* db = (uint4*)(Bs[buf] + bn_ld * LDS_STRIDE + bk_ld);
        db[0] = uB[0]; db[1] = uB[1];
#endif
        wait_async0();   // async tiles must have landed before the barrier
    };

    const int nk = Ktot / BK;
    loadTiles(0, 0);
    storeTiles(0);
    __syncthreads();

    for (int i = 0; i < nk; ++i) {
        const bool more = (i + 1) < nk;
        const int cur = i & 1;
        if (more) loadTiles((i + 1) * BK, cur ^ 1);   // overlap with WMMAs

        // ---- A fragments (two 16x32 tiles) ----
        // lane<16: M=l16, K {0..7}U{16..23}; lane>=16: M=l16, K {8..15}U{24..31}
        union { uint4 q[2]; v16bf v; } af[2];
#pragma unroll
        for (int s = 0; s < 2; ++s) {
            const u16* arow = As[cur] + (wm + s * 16 + l16) * LDS_STRIDE + half * 8;
            af[s].q[0] = *(const uint4*)(arow);
            af[s].q[1] = *(const uint4*)(arow + 16);
        }
        // ---- B fragments (four 32x16 tiles) ----
        union { uint4 q[2]; v16bf v; } bfr[4];
#pragma unroll
        for (int t = 0; t < 4; ++t) {
            const u16* brow = Bs[cur] + (wn + t * 16 + l16) * LDS_STRIDE + half * 8;
            bfr[t].q[0] = *(const uint4*)(brow);
            bfr[t].q[1] = *(const uint4*)(brow + 16);
        }
        // ---- 8 back-to-back WMMAs ----
#pragma unroll
        for (int s = 0; s < 2; ++s)
#pragma unroll
            for (int t = 0; t < 4; ++t)
                acc[s][t] = __builtin_amdgcn_wmma_f32_16x16x32_bf16(
                    false, af[s].v, false, bfr[t].v, (short)0, acc[s][t],
                    false, false);

        if (more) storeTiles(cur ^ 1);
        __syncthreads();
    }

    // ---- epilogue: D layout -> reg r, lanes 0-15: (M=r,N=lane); 16-31: M=r+8
#pragma unroll
    for (int s = 0; s < 2; ++s) {
#pragma unroll
        for (int t = 0; t < 4; ++t) {
#pragma unroll
            for (int r = 0; r < 8; ++r) {
                int gm = m0 + wm + s * 16 + r + half * 8;
                int gn = n0 + wn + t * 16 + l16;
                float v = acc[s][t][r];
                if (bias) v += bias[gn];
                if (relu) v = fmaxf(v, 0.0f);
                size_t idx = (size_t)gm * Ntot + gn;
                if (Cf) Cf[idx] = v;
                if (Cb) Cb[idx] = f2bf(v);
            }
        }
    }
}

// --------------------------- transpose kernels -----------------------------
// out[C][R] = in[R][C]; 32x32 tiles through LDS, block (32,8).

__global__ __launch_bounds__(256)
void transpose_f32_to_bf16(const float* __restrict__ in, u16* __restrict__ out,
                           int R, int C) {
    __shared__ u16 tile[32][33];
    int bx = blockIdx.x * 32;   // col base (in)
    int by = blockIdx.y * 32;   // row base (in)
#pragma unroll
    for (int i = 0; i < 4; ++i) {
        int r = by + threadIdx.y + i * 8;
        tile[threadIdx.y + i * 8][threadIdx.x] =
            f2bf(in[(size_t)r * C + bx + threadIdx.x]);
    }
    __syncthreads();
#pragma unroll
    for (int i = 0; i < 4; ++i) {
        int c = bx + threadIdx.y + i * 8;
        out[(size_t)c * R + by + threadIdx.x] = tile[threadIdx.x][threadIdx.y + i * 8];
    }
}

__global__ __launch_bounds__(256)
void transpose_bf16(const u16* __restrict__ in, u16* __restrict__ out,
                    int R, int C) {
    __shared__ u16 tile[32][33];
    int bx = blockIdx.x * 32;
    int by = blockIdx.y * 32;
#pragma unroll
    for (int i = 0; i < 4; ++i) {
        int r = by + threadIdx.y + i * 8;
        tile[threadIdx.y + i * 8][threadIdx.x] = in[(size_t)r * C + bx + threadIdx.x];
    }
    __syncthreads();
#pragma unroll
    for (int i = 0; i < 4; ++i) {
        int c = bx + threadIdx.y + i * 8;
        out[(size_t)c * R + by + threadIdx.x] = tile[threadIdx.x][threadIdx.y + i * 8];
    }
}

__global__ void convert_f32_to_bf16(const float* __restrict__ in,
                                    u16* __restrict__ out, int n) {
    int i = blockIdx.x * blockDim.x + threadIdx.x;
    if (i < n) out[i] = f2bf(in[i]);
}

// --------------------------- small fused kernels ---------------------------

// vals[row] = dot(X[row,:ncols], w) + b[0]    (one wave per row)
__global__ __launch_bounds__(256)
void rowdot(const float* __restrict__ X, const float* __restrict__ w,
            const float* __restrict__ b, float* __restrict__ out, int ncols) {
    int row  = blockIdx.x * 8 + (threadIdx.x >> 5);
    int lane = threadIdx.x & 31;
    const float* xr = X + (size_t)row * ncols;
    float s = 0.0f;
    for (int c = lane; c < ncols; c += 32) s += xr[c] * w[c];
    s = wave_sum(s);
    if (lane == 0) out[row] = s + b[0];
}

// gh[j] = dot(W_hh[j,:], h) + b_hh[j]   for j in [0,1536)  (one wave per j)
__global__ __launch_bounds__(256)
void gh_kernel(const float* __restrict__ W_hh, const float* __restrict__ b_hh,
               const float* __restrict__ h, float* __restrict__ gh) {
    int j    = blockIdx.x * 8 + (threadIdx.x >> 5);
    int lane = threadIdx.x & 31;
    const float* wr = W_hh + (size_t)j * HH;
    float s = 0.0f;
    for (int k = lane; k < HH; k += 32) s += wr[k] * h[k];
    s = wave_sum(s);
    if (lane == 0) gh[j] = s + b_hh[j];
}

// One GRU step fused with the w_out2 dot (one wave per graph node).
__global__ __launch_bounds__(256)
void step_vals(const float* __restrict__ GI, const float* __restrict__ gh,
               const float* __restrict__ h, const float* __restrict__ w2,
               const float* __restrict__ b2, float* __restrict__ vals) {
    int row  = blockIdx.x * 8 + (threadIdx.x >> 5);
    int lane = threadIdx.x & 31;
    const float* gi = GI + (size_t)row * H3;
    float s = 0.0f;
    for (int j = lane; j < HH; j += 32) {
        float r  = sigm(gi[j] + gh[j]);
        float z  = sigm(gi[HH + j] + gh[HH + j]);
        float ng = tanhf(gi[2 * HH + j] + r * gh[2 * HH + j]);
        s += ((1.0f - z) * ng + z * h[j]) * w2[j];
    }
    s = wave_sum(s);
    if (lane == 0) vals[row] = s + b2[0];
}

// ------------------------------- argmax ------------------------------------

__global__ __launch_bounds__(256)
void argmax_stage1(const float* __restrict__ vals, const int* __restrict__ mask,
                   float* __restrict__ pv, int* __restrict__ pi) {
    __shared__ float sv[256];
    __shared__ int   si[256];
    int tid = threadIdx.x;
    int i   = blockIdx.x * 256 + tid;
    sv[tid] = mask[i] ? vals[i] : -1e30f;
    si[tid] = i;
    __syncthreads();
    for (int s = 128; s > 0; s >>= 1) {
        if (tid < s) {
            if (sv[tid + s] > sv[tid] ||
                (sv[tid + s] == sv[tid] && si[tid + s] < si[tid])) {
                sv[tid] = sv[tid + s]; si[tid] = si[tid + s];
            }
        }
        __syncthreads();
    }
    if (tid == 0) { pv[blockIdx.x] = sv[0]; pi[blockIdx.x] = si[0]; }
}

__global__ void argmax_stage2(const float* __restrict__ pv,
                              const int* __restrict__ pi, int nparts,
                              int* __restrict__ a_out) {
    __shared__ float sv[32];
    __shared__ int   si[32];
    int tid = threadIdx.x;
    sv[tid] = (tid < nparts) ? pv[tid] : -1e30f;
    si[tid] = (tid < nparts) ? pi[tid] : 0x7FFFFFFF;
    __syncthreads();
    for (int s = 16; s > 0; s >>= 1) {
        if (tid < s) {
            if (sv[tid + s] > sv[tid] ||
                (sv[tid + s] == sv[tid] && si[tid + s] < si[tid])) {
                sv[tid] = sv[tid + s]; si[tid] = si[tid + s];
            }
        }
        __syncthreads();
    }
    if (tid == 0) *a_out = si[0];
}

// ---------------------------- bookkeeping ----------------------------------

__global__ void init_mask(const unsigned char* __restrict__ pm,
                          int* __restrict__ mask) {
    int i = blockIdx.x * blockDim.x + threadIdx.x;
    if (i < NN) mask[i] = pm[i] ? 1 : 0;
}

// After first argmax: record action/q, kill mask entry, h = G[a,:]
__global__ void post0(const int* __restrict__ a_ptr, const float* __restrict__ vals,
                      int* __restrict__ mask, const float* __restrict__ G,
                      float* __restrict__ h, float* __restrict__ out) {
    int a = *a_ptr;
    int j = threadIdx.x;
    if (j == 0) { mask[a] = 0; out[0] = (float)a; out[BSZ] = vals[a]; }
    h[j] = G[(size_t)a * HH + j];
}

// After step-t argmax: record, kill mask entry, recompute GRU row a -> new h
__global__ void post_t(const int* __restrict__ a_ptr, const float* __restrict__ vals,
                       int* __restrict__ mask, const float* __restrict__ GI,
                       const float* __restrict__ gh, float* __restrict__ h,
                       float* __restrict__ out, int t) {
    int a = *a_ptr;
    int j = threadIdx.x;
    if (j == 0) { mask[a] = 0; out[t] = (float)a; out[BSZ + t] = vals[a]; }
    const float* gi = GI + (size_t)a * H3;
    float r  = sigm(gi[j] + gh[j]);
    float z  = sigm(gi[HH + j] + gh[HH + j]);
    float ng = tanhf(gi[2 * HH + j] + r * gh[2 * HH + j]);
    h[j] = (1.0f - z) * ng + z * h[j];   // each thread touches only its own j
}

// ------------------------------ launcher -----------------------------------

static inline size_t align256(size_t x) { return (x + 255) & ~(size_t)255; }

extern "C" void kernel_launch(void* const* d_in, const int* in_sizes, int n_in,
                              void* d_out, int out_size, void* d_ws, size_t ws_size,
                              hipStream_t stream) {
    (void)in_sizes; (void)n_in; (void)out_size; (void)ws_size;

    const float* state  = (const float*)d_in[0];           // [N,D]
    const float* adj    = (const float*)d_in[1];           // [N,N]
    const unsigned char* pmask = (const unsigned char*)d_in[2];
    const float* W1     = (const float*)d_in[3];           // [D,H]
    const float* b1     = (const float*)d_in[4];
    const float* W2     = (const float*)d_in[5];           // [H,H]
    const float* b2     = (const float*)d_in[6];
    const float* w_out1 = (const float*)d_in[7];
    const float* b_out1 = (const float*)d_in[8];
    const float* w_out2 = (const float*)d_in[9];
    const float* b_out2 = (const float*)d_in[10];
    const float* W_ih   = (const float*)d_in[11];          // [3H,H]
    const float* W_hh   = (const float*)d_in[12];          // [3H,H]
    const float* b_ih   = (const float*)d_in[13];
    const float* b_hh   = (const float*)d_in[14];
    float* out = (float*)d_out;                            // [20 actions | 20 q]

    // ---- workspace carve-up ----
    char* ws = (char*)d_ws;
    size_t off = 0;
    auto take = [&](size_t bytes) { char* p = ws + off; off = align256(off + bytes); return p; };
    u16*   W1t  = (u16*)  take((size_t)HH * DD * 2);
    u16*   W2t  = (u16*)  take((size_t)HH * HH * 2);
    u16*   Wihb = (u16*)  take((size_t)H3 * HH * 2);
    u16*   X1b  = (u16*)  take((size_t)NN * HH * 2);
    u16*   X1t  = (u16*)  take((size_t)HH * NN * 2);
    u16*   F1b  = (u16*)  take((size_t)NN * HH * 2);
    u16*   X2b  = (u16*)  take((size_t)NN * HH * 2);
    u16*   X2t  = (u16*)  take((size_t)HH * NN * 2);
    float* G    = (float*)take((size_t)NN * HH * 4);
    u16*   Gb   = (u16*)  take((size_t)NN * HH * 2);
    float* GI   = (float*)take((size_t)NN * H3 * 4);
    float* vals = (float*)take((size_t)NN * 4);
    int*   mask = (int*)  take((size_t)NN * 4);
    float* gh   = (float*)take((size_t)H3 * 4);
    float* hrow = (float*)take((size_t)HH * 4);
    float* pv   = (float*)take(32 * 4);
    int*   pi   = (int*)  take(32 * 4);
    int*   aidx = (int*)  take(4);

    // ---- prep: weight transposes / conversions ----
    transpose_f32_to_bf16<<<dim3(HH / 32, DD / 32), dim3(32, 8), 0, stream>>>(W1, W1t, DD, HH);
    transpose_f32_to_bf16<<<dim3(HH / 32, HH / 32), dim3(32, 8), 0, stream>>>(W2, W2t, HH, HH);
    convert_f32_to_bf16<<<(H3 * HH + 255) / 256, 256, 0, stream>>>(W_ih, Wihb, H3 * HH);

    // ---- GCN chain ----
    dim3 g512(HH / BN, NN / BM), g1536(H3 / BN, NN / BM);
    // X1 = state @ W1              (bf16 out)
    gemm_bf16_wmma<<<g512, 256, 0, stream>>>(state, nullptr, W1t,
                                             nullptr, X1b, nullptr, 0, NN, HH, DD);
    transpose_bf16<<<dim3(HH / 32, NN / 32), dim3(32, 8), 0, stream>>>(X1b, X1t, NN, HH);
    // F1 = relu(adj @ X1 + b1)     (bf16 out)
    gemm_bf16_wmma<<<g512, 256, 0, stream>>>(adj, nullptr, X1t,
                                             nullptr, F1b, b1, 1, NN, HH, NN);
    // X2 = F1 @ W2                 (bf16 out)
    gemm_bf16_wmma<<<g512, 256, 0, stream>>>(nullptr, F1b, W2t,
                                             nullptr, X2b, nullptr, 0, NN, HH, HH);
    transpose_bf16<<<dim3(HH / 32, NN / 32), dim3(32, 8), 0, stream>>>(X2b, X2t, NN, HH);
    // G = relu(adj @ X2 + b2)      (fp32 + bf16 out)
    gemm_bf16_wmma<<<g512, 256, 0, stream>>>(adj, nullptr, X2t,
                                             G, Gb, b2, 1, NN, HH, NN);
    // GI = G @ W_ih^T + b_ih       (loop-invariant GRU input, fp32 out)
    gemm_bf16_wmma<<<g1536, 256, 0, stream>>>(nullptr, Gb, Wihb,
                                              GI, nullptr, b_ih, 0, NN, H3, HH);

    // ---- selection chain ----
    rowdot<<<NN / 8, 256, 0, stream>>>(G, w_out1, b_out1, vals, HH);
    init_mask<<<NN / 256, 256, 0, stream>>>(pmask, mask);
    argmax_stage1<<<NN / 256, 256, 0, stream>>>(vals, mask, pv, pi);
    argmax_stage2<<<1, 32, 0, stream>>>(pv, pi, NN / 256, aidx);
    post0<<<1, HH, 0, stream>>>(aidx, vals, mask, G, hrow, out);

    for (int t = 1; t < BSZ; ++t) {
        gh_kernel<<<H3 / 8, 256, 0, stream>>>(W_hh, b_hh, hrow, gh);
        step_vals<<<NN / 8, 256, 0, stream>>>(GI, gh, hrow, w_out2, b_out2, vals);
        argmax_stage1<<<NN / 256, 256, 0, stream>>>(vals, mask, pv, pi);
        argmax_stage2<<<1, 32, 0, stream>>>(pv, pi, NN / 256, aidx);
        post_t<<<1, HH, 0, stream>>>(aidx, vals, mask, GI, gh, hrow, out, t);
    }
}
